// LabelAttentionClassifier_20280835572263
// MI455X (gfx1250) — matrix operand order repaired and checked by reading, not compiled, not run on hardware
//
#include <hip/hip_runtime.h>
#include <hip/hip_bf16.h>
#include <math.h>

#define B_    8
#define S_    512
#define D_    768
#define NC    8921
#define NPAD  8928          // 558 * 16
#define NT    (NPAD / 16)   // 558 n-tiles
#define WSTRIDE 516         // LDS row stride (floats), skewed for bank-conflict-free access

typedef __attribute__((ext_vector_type(16))) __bf16 v16bf;
typedef __attribute__((ext_vector_type(8)))  float  v8f;

union Frag16 {
    v16bf v;
    uint4 q[2];
    unsigned int u[8];
};

__device__ __forceinline__ unsigned short f2bf(float f) {
    unsigned int u = __builtin_bit_cast(unsigned int, f);
    unsigned int r = u + 0x7FFFu + ((u >> 16) & 1u);   // round-to-nearest-even
    return (unsigned short)(r >> 16);
}
__device__ __forceinline__ unsigned int pack2bf(float lo, float hi) {
    return (unsigned int)f2bf(lo) | ((unsigned int)f2bf(hi) << 16);
}

// ---------------------------------------------------------------------------
// Prep: fp32 -> bf16 copies. Ebf row-major [B,S,D] (GEMM1 B-frags),
// Etbf transposed [B,D,S] (GEMM2 B-frags), Qbf zero-padded [NPAD,D].
// ---------------------------------------------------------------------------
__global__ void cvt_e_kernel(const float* __restrict__ E,
                             unsigned short* __restrict__ Ebf,
                             unsigned short* __restrict__ Etbf) {
    size_t idx = (size_t)blockIdx.x * blockDim.x + threadIdx.x;
    if (idx >= (size_t)B_ * S_ * D_) return;
    unsigned short h = f2bf(E[idx]);
    Ebf[idx] = h;
    int d    = (int)(idx % D_);
    size_t r = idx / D_;
    int s    = (int)(r % S_);
    int b    = (int)(r / S_);
    Etbf[((size_t)b * D_ + d) * S_ + s] = h;
}

__global__ void cvt_q_kernel(const float* __restrict__ Q,
                             unsigned short* __restrict__ Qbf) {
    size_t idx = (size_t)blockIdx.x * blockDim.x + threadIdx.x;
    if (idx >= (size_t)NPAD * D_) return;
    int n = (int)(idx / D_);
    Qbf[idx] = (n < NC) ? f2bf(Q[idx]) : (unsigned short)0;
}

// ---------------------------------------------------------------------------
// Fused: GEMM1 (WMMA bf16) -> LDS -> masked softmax -> attention output,
//        GEMM2 (WMMA bf16 from LDS weights) fused with classification dot.
// One workgroup (8 wave32) per (batch b, 16-row n-tile).
// ---------------------------------------------------------------------------
__global__ __launch_bounds__(256)
void la_fused_kernel(const unsigned short* __restrict__ Ebf,
                     const unsigned short* __restrict__ Etbf,
                     const unsigned short* __restrict__ Qbf,
                     const unsigned char*  __restrict__ mask,
                     const float* __restrict__ cls,
                     float* __restrict__ out) {
    __shared__ float ldsW[16][WSTRIDE];   // 16 n-rows x 512 s (skewed)
    __shared__ float ldsLogit[16];

    const int tid  = threadIdx.x;
    const int wave = tid >> 5;
    const int lane = tid & 31;
    const int lo   = lane & 15;     // row/col within 16
    const int hi   = lane >> 4;     // half-wave select
    const int n0   = blockIdx.x * 16;
    const int b    = blockIdx.y;

    if (tid < 16) ldsLogit[tid] = 0.0f;

    // ---------------- Phase 1: logits = Q * E^T / sqrt(D) -----------------
    // A-frag (16x32 bf16): lane holds row n0+lo, K halves split by hi.
    const unsigned short* qrow = Qbf + (size_t)(n0 + lo) * D_ + hi * 8;
    const unsigned short* erow[4];
#pragma unroll
    for (int i = 0; i < 4; ++i) {
        int s = (wave * 4 + i) * 16 + lo;
        erow[i] = Ebf + (size_t)(b * S_ + s) * D_ + hi * 8;
    }
    v8f acc1[4] = {};
    for (int k = 0; k < D_ / 32; ++k) {
        const int d0 = k * 32;
        Frag16 a;
        a.q[0] = *(const uint4*)(qrow + d0);        // K = koff .. koff+7
        a.q[1] = *(const uint4*)(qrow + d0 + 16);   // K = 16+koff .. 23+koff
        if (k + 1 < D_ / 32) {
            __builtin_prefetch(qrow + d0 + 32, 0, 3);
            __builtin_prefetch(erow[0] + d0 + 32, 0, 3);
        }
#pragma unroll
        for (int i = 0; i < 4; ++i) {
            Frag16 bf;
            bf.q[0] = *(const uint4*)(erow[i] + d0);      // K = hi*8 + 0..7
            bf.q[1] = *(const uint4*)(erow[i] + d0 + 16); // K = 16 + hi*8 + 0..7
            acc1[i] = __builtin_amdgcn_wmma_f32_16x16x32_bf16(
                false, a.v, false, bf.v, (short)0, acc1[i], false, false);
        }
    }
    const float scale = 0.03608439182435161f;  // 1/sqrt(768)
#pragma unroll
    for (int i = 0; i < 4; ++i) {
        int sc = (wave * 4 + i) * 16 + lo;
#pragma unroll
        for (int r = 0; r < 8; ++r)
            ldsW[hi * 8 + r][sc] = acc1[i][r] * scale;
    }
    __syncthreads();

    // ---------------- Phase 2: masked softmax over S, emit weights --------
    {
        const int r2 = tid >> 4;          // n-row 0..15
        const int c0 = (tid & 15) * 32;   // 32 s-columns per thread
        const unsigned char* mrow = mask + (size_t)b * S_;
        float tmp[32];
        float mx = -INFINITY;
#pragma unroll
        for (int j = 0; j < 32; ++j) {
            int s = c0 + j;
            float v = ldsW[r2][s];
            v = mrow[s] ? v : -INFINITY;
            tmp[j] = v;
            mx = fmaxf(mx, v);
        }
#pragma unroll
        for (int off = 1; off < 16; off <<= 1)
            mx = fmaxf(mx, __shfl_xor(mx, off, 16));
        float sum = 0.0f;
#pragma unroll
        for (int j = 0; j < 32; ++j) {
            float e = __expf(tmp[j] - mx);  // exp(-inf)=0 handles the mask
            tmp[j] = e;
            sum += e;
        }
#pragma unroll
        for (int off = 1; off < 16; off <<= 1)
            sum += __shfl_xor(sum, off, 16);
        const float inv = 1.0f / sum;
        const int nrow = n0 + r2;
        float* abase = out + (size_t)B_ * NC + ((size_t)b * NC + nrow) * S_ + c0;
#pragma unroll
        for (int j = 0; j < 32; j += 4) {
            float4 w4;
            w4.x = tmp[j + 0] * inv;
            w4.y = tmp[j + 1] * inv;
            w4.z = tmp[j + 2] * inv;
            w4.w = tmp[j + 3] * inv;
            *(float4*)&ldsW[r2][c0 + j] = w4;
            if (nrow < NC) *(float4*)(abase + j) = w4;   // attention_weights out
        }
    }
    __syncthreads();

    // ---------------- Phase 3: code_emb = W * E, fused cls dot ------------
    float part[8] = {};
    v8f acc2[6] = {};
    const int koffA = hi * 8;
    const float* wrow = &ldsW[lo][0];
    const unsigned short* etrow[6];
#pragma unroll
    for (int t = 0; t < 6; ++t) {
        int d = (wave * 6 + t) * 16 + lo;
        etrow[t] = Etbf + (size_t)(b * D_ + d) * S_ + hi * 8;
    }
    for (int kk = 0; kk < S_ / 32; ++kk) {
        const int sb = kk * 32;
        float4 f0 = *(const float4*)(wrow + sb + koffA);
        float4 f1 = *(const float4*)(wrow + sb + koffA + 4);
        float4 f2 = *(const float4*)(wrow + sb + 16 + koffA);
        float4 f3 = *(const float4*)(wrow + sb + 16 + koffA + 4);
        Frag16 a;
        a.u[0] = pack2bf(f0.x, f0.y);
        a.u[1] = pack2bf(f0.z, f0.w);
        a.u[2] = pack2bf(f1.x, f1.y);
        a.u[3] = pack2bf(f1.z, f1.w);
        a.u[4] = pack2bf(f2.x, f2.y);
        a.u[5] = pack2bf(f2.z, f2.w);
        a.u[6] = pack2bf(f3.x, f3.y);
        a.u[7] = pack2bf(f3.z, f3.w);
#pragma unroll
        for (int t = 0; t < 6; ++t) {
            Frag16 bf;
            bf.q[0] = *(const uint4*)(etrow[t] + sb);
            bf.q[1] = *(const uint4*)(etrow[t] + sb + 16);
            acc2[t] = __builtin_amdgcn_wmma_f32_16x16x32_bf16(
                false, a.v, false, bf.v, (short)0, acc2[t], false, false);
        }
    }
#pragma unroll
    for (int t = 0; t < 6; ++t) {
        int d = (wave * 6 + t) * 16 + lo;
#pragma unroll
        for (int r = 0; r < 8; ++r) {
            int n = n0 + hi * 8 + r;
            float cw = (n < NC) ? cls[(size_t)n * D_ + d] : 0.0f;
            part[r] += acc2[t][r] * cw;
        }
    }
#pragma unroll
    for (int r = 0; r < 8; ++r) {
        float p = part[r];
#pragma unroll
        for (int off = 1; off < 16; off <<= 1)
            p += __shfl_xor(p, off, 16);
        if (lo == 0) atomicAdd(&ldsLogit[hi * 8 + r], p);  // ds_add_f32
    }
    __syncthreads();
    if (tid < 16) {
        int n = n0 + tid;
        if (n < NC) out[(size_t)b * NC + n] = ldsLogit[tid];
    }
}

// ---------------------------------------------------------------------------
extern "C" void kernel_launch(void* const* d_in, const int* in_sizes, int n_in,
                              void* d_out, int out_size, void* d_ws, size_t ws_size,
                              hipStream_t stream) {
    (void)in_sizes; (void)n_in; (void)out_size; (void)ws_size;
    const float*         E    = (const float*)d_in[0];
    const unsigned char* mask = (const unsigned char*)d_in[1];
    const float*         Q    = (const float*)d_in[2];
    const float*         cls  = (const float*)d_in[3];
    float*               out  = (float*)d_out;

    unsigned short* Ebf  = (unsigned short*)d_ws;
    unsigned short* Etbf = Ebf  + (size_t)B_ * S_ * D_;
    unsigned short* Qbf  = Etbf + (size_t)B_ * S_ * D_;

    const int tot_e = B_ * S_ * D_;
    cvt_e_kernel<<<(tot_e + 255) / 256, 256, 0, stream>>>(E, Ebf, Etbf);
    const int tot_q = NPAD * D_;
    cvt_q_kernel<<<(tot_q + 255) / 256, 256, 0, stream>>>(Q, Qbf);

    la_fused_kernel<<<dim3(NT, B_), 256, 0, stream>>>(Ebf, Etbf, Qbf, mask, cls, out);
}
